// DeepseekV2Attention_51488067944633
// MI455X (gfx1250) — matrix operand order repaired and checked by reading, not compiled, not run on hardware
//
#include <hip/hip_runtime.h>
#include <hip/hip_bf16.h>

// ---------------------------------------------------------------------------
// DeepSeek-V2 MLA attention forward for MI455X (gfx1250), bf16 WMMA pipeline
// with CDNA5 async global->LDS staging (GLOBAL_LOAD_ASYNC_TO_LDS_B128).
// B=2, S=2048, H=2048, NH=16, D_NOPE=128, D_ROPE=64, D_V=128, LORA=512.
// Attention uses only the nope dims (rope of q is dead code in the reference).
// ---------------------------------------------------------------------------

typedef __attribute__((ext_vector_type(16))) __bf16 v16bf;
typedef __attribute__((ext_vector_type(8)))  float  v8f;

__device__ inline __bf16 f2bf(float f) {
    union { unsigned u; float f; } cv; cv.f = f;
    unsigned r = cv.u + 0x7fffu + ((cv.u >> 16) & 1u);   // round-to-nearest-even
    unsigned short h = (unsigned short)(r >> 16);
    __bf16 b; __builtin_memcpy(&b, &h, 2); return b;
}

__device__ inline v8f v8f_zero() {
    v8f z = {0.f,0.f,0.f,0.f,0.f,0.f,0.f,0.f};
    return z;
}

// CDNA5 async copy: 16B global -> LDS, tracked by ASYNCcnt (ISA 15.18.3 op 98).
// VDST is the LDS byte offset; flat-address low 32 bits == LDS offset per the
// aperture spec.
__device__ __forceinline__ void async_copy_b128(const void* gptr, void* lds_ptr) {
    uint32_t lds_off = (uint32_t)(uintptr_t)lds_ptr;
    asm volatile("global_load_async_to_lds_b128 %0, %1, off"
                 :: "v"(lds_off), "v"(gptr)
                 : "memory");
}
__device__ __forceinline__ void async_wait0() {
    asm volatile("s_wait_asynccnt 0" ::: "memory");
}

// ---------------------------------------------------------------------------
// fp32 -> bf16 conversion kernels
// ---------------------------------------------------------------------------
__global__ __launch_bounds__(256) void f32_to_bf16_kernel(
        const float* __restrict__ src, __bf16* __restrict__ dst, int n) {
    int i = blockIdx.x * 256 + threadIdx.x;
    if (i < n) dst[i] = f2bf(src[i]);
}

// gather only nope columns of wq: dst[K=2048][N=2048], col n -> wq col (n/128)*192 + n%128
__global__ __launch_bounds__(256) void wq_gather_bf16_kernel(
        const float* __restrict__ wq, __bf16* __restrict__ dst) {
    int i = blockIdx.x * 256 + threadIdx.x;           // 0 .. 2048*2048-1
    int r = i >> 11;
    int n = i & 2047;
    dst[i] = f2bf(wq[(size_t)r * 3072 + (n >> 7) * 192 + (n & 127)]);
}

// ---------------------------------------------------------------------------
// Tiled bf16 GEMM, v_wmma_f32_16x16x32_bf16, async staging, double-buffered.
// Block tile 128xBN (BN = 64 or 128), BK=32, 8 waves.
// Wave tile 32 x (BN/2): 2 M-subtiles x NS N-subtiles, NS = BN/32.
// A: MxK row-major bf16, B: KxN row-major bf16, C: f32 or bf16.
// All dims assumed multiples of the tile sizes (true for this problem).
// ---------------------------------------------------------------------------
template<int BN, bool BF16_OUT>
__global__ __launch_bounds__(256) void gemm_bf16_kernel(
        const __bf16* __restrict__ A, const __bf16* __restrict__ B,
        void* __restrict__ C, int K, int lda, int ldb, int ldc) {
    constexpr int BM = 128, BK = 32;
    constexpr int NS = BN / 32;                // N-subtiles per wave
    __shared__ __bf16 As[2][BM][BK + 8];       // stride 80B (16B aligned)
    __shared__ __bf16 Bs[2][BK][BN + 8];       // stride 144B/272B (16B aligned)

    const int tid  = threadIdx.x;
    const int wave = tid >> 5;
    const int lane = tid & 31;
    const int lr   = lane & 15;
    const int lh   = lane >> 4;
    const int wm   = wave >> 1;                // 0..3 -> M offset wm*32
    const int wn   = wave & 1;                 // 0..1 -> N offset wn*(BN/2)
    const int bm   = blockIdx.y * BM;
    const int bn   = blockIdx.x * BN;

    auto stage = [&](int buf, int kb) {
        // A tile 128x32
        #pragma unroll
        for (int it = 0; it < 2; ++it) {
            int idx = tid + it * 256;          // 0..511 chunks of 8
            int row = idx >> 2;
            int col = (idx & 3) << 3;
            async_copy_b128(A + (size_t)(bm + row) * lda + kb + col,
                            &As[buf][row][col]);
        }
        // B tile 32xBN
        #pragma unroll
        for (int it = 0; it < BN / 64; ++it) {
            int idx = tid + it * 256;
            int row = idx / (BN / 8);
            int col = (idx % (BN / 8)) << 3;
            async_copy_b128(B + (size_t)(kb + row) * ldb + bn + col,
                            &Bs[buf][row][col]);
        }
    };

    v8f acc[2][NS];
    #pragma unroll
    for (int i = 0; i < 2; ++i)
        #pragma unroll
        for (int j = 0; j < NS; ++j) acc[i][j] = v8f_zero();

    stage(0, 0);
    async_wait0();
    __syncthreads();

    int cur = 0;
    for (int kb = 0; kb < K; kb += BK) {
        if (kb + BK < K) stage(cur ^ 1, kb + BK);   // overlap with WMMA below

        // fragments per ISA 7.12.2 layouts
        v16bf af[2], bfr[NS];
        #pragma unroll
        for (int i = 0; i < 2; ++i) {
            int mrow = wm * 32 + i * 16 + lr;
            #pragma unroll
            for (int v = 0; v < 8; ++v) {
                int k0 = (v < 4) ? (2 * v + 8 * lh) : (16 + 2 * (v - 4) + 8 * lh);
                af[i][2 * v]     = As[cur][mrow][k0];
                af[i][2 * v + 1] = As[cur][mrow][k0 + 1];
            }
        }
        #pragma unroll
        for (int j = 0; j < NS; ++j) {
            int ncol = wn * (BN / 2) + j * 16 + lr;
            #pragma unroll
            for (int v = 0; v < 8; ++v) {
                int k0 = (v < 4) ? (2 * v + 8 * lh) : (16 + 2 * (v - 4) + 8 * lh);
                bfr[j][2 * v]     = Bs[cur][k0][ncol];
                bfr[j][2 * v + 1] = Bs[cur][k0 + 1][ncol];
            }
        }
        #pragma unroll
        for (int i = 0; i < 2; ++i)
            #pragma unroll
            for (int j = 0; j < NS; ++j)
                acc[i][j] = __builtin_amdgcn_wmma_f32_16x16x32_bf16(
                    false, af[i], false, bfr[j], (short)0, acc[i][j], false, false);

        async_wait0();        // next tile fully landed in LDS
        __syncthreads();
        cur ^= 1;
    }

    // store C (C layout: lane -> col lr, VGPR v -> row v + 8*lh)
    #pragma unroll
    for (int i = 0; i < 2; ++i) {
        #pragma unroll
        for (int j = 0; j < NS; ++j) {
            int n = bn + wn * (BN / 2) + j * 16 + lr;
            #pragma unroll
            for (int v = 0; v < 8; ++v) {
                int m = bm + wm * 32 + i * 16 + v + 8 * lh;
                float val = acc[i][j][v];
                if constexpr (BF16_OUT)
                    ((__bf16*)C)[(size_t)m * ldc + n] = f2bf(val);
                else
                    ((float*)C)[(size_t)m * ldc + n] = val;
            }
        }
    }
}

// ---------------------------------------------------------------------------
// RMSNorm(ckv[:,:512]) + rope(k_pe) -> kv_fused (f32) and compressed bf16.
// One 256-thread block per (b,s) row.
// ---------------------------------------------------------------------------
__global__ __launch_bounds__(256) void rmsnorm_rope_kernel(
        const float* __restrict__ ckv, const float* __restrict__ ln_w,
        const int* __restrict__ positions,
        float* __restrict__ kvf, __bf16* __restrict__ comp_bf) {
    const int row = blockIdx.x;
    const int tid = threadIdx.x;
    const float* x = ckv + (size_t)row * 576;

    float v0 = x[tid], v1 = x[tid + 256];
    float ss = v0 * v0 + v1 * v1;
    #pragma unroll
    for (int off = 16; off; off >>= 1) ss += __shfl_xor(ss, off, 32);

    __shared__ float red[8];
    __shared__ float rsv;
    if ((tid & 31) == 0) red[tid >> 5] = ss;
    __syncthreads();
    if (tid == 0) {
        float t = 0.f;
        #pragma unroll
        for (int i = 0; i < 8; ++i) t += red[i];
        rsv = rsqrtf(t * (1.0f / 512.0f) + 1e-6f);
    }
    __syncthreads();
    const float rs = rsv;

    float y0 = v0 * rs * ln_w[tid];
    float y1 = v1 * rs * ln_w[tid + 256];
    kvf[(size_t)row * 576 + tid]       = y0;
    kvf[(size_t)row * 576 + tid + 256] = y1;
    comp_bf[(size_t)row * 512 + tid]       = f2bf(y0);
    comp_bf[(size_t)row * 512 + tid + 256] = f2bf(y1);

    if (tid < 32) {   // rope the 64-dim k_pe (half = 32)
        float pos = (float)positions[row];
        float inv_freq = __expf(-((float)tid) * (13.815510557964274f / 32.0f)); // ln(1e6)
        float ang = pos * inv_freq;
        float s, c;
        __sincosf(ang, &s, &c);
        float x1 = x[512 + tid], x2 = x[544 + tid];
        kvf[(size_t)row * 576 + 512 + tid] = x1 * c - x2 * s;
        kvf[(size_t)row * 576 + 544 + tid] = x2 * c + x1 * s;
    }
}

// ---------------------------------------------------------------------------
// Flash attention per (b, h, 128-row Q block). Causal, scaling=1/sqrt(128).
// Q: (B*S, 2048) bf16 (head h at col h*128).
// KV: (B*S, 4096) bf16 (head h: K at h*256, V at h*256+128).
// Oa: (B*S, 2048) bf16 attention output (pre out-proj).
// ---------------------------------------------------------------------------
__global__ __launch_bounds__(256) void mla_attn_kernel(
        const __bf16* __restrict__ Q, const __bf16* __restrict__ KV,
        __bf16* __restrict__ Oa) {
    constexpr int S = 2048, BQ = 128, BKV = 64, D = 128;
    const float scaling = 0.08838834764831845f;   // 128^-0.5

    const int qblk  = blockIdx.x;
    const int b     = blockIdx.y >> 4;
    const int h     = blockIdx.y & 15;
    const int qbase = qblk * BQ;

    const int tid   = threadIdx.x;
    const int wave  = tid >> 5;
    const int lane  = tid & 31;
    const int lr    = lane & 15;
    const int lh    = lane >> 4;
    const int mbase = wave * 16;       // 8 waves x 16 rows = 128

    __shared__ __bf16 Ks[BKV][D + 8];
    __shared__ __bf16 Vs[BKV][D + 8];
    __shared__ __bf16 Ps[8][16][BKV + 8];   // per-wave P re-layout patch

    // preload Q A-fragments for 4 K-steps of 32 (stay in registers all loop)
    v16bf qf[4];
    const size_t qrowbase = (size_t)(b * S + qbase + mbase + lr) * 2048 + h * 128;
    #pragma unroll
    for (int kk = 0; kk < 4; ++kk) {
        #pragma unroll
        for (int v = 0; v < 8; ++v) {
            int k0 = kk * 32 + ((v < 4) ? (2 * v + 8 * lh) : (16 + 2 * (v - 4) + 8 * lh));
            qf[kk][2 * v]     = Q[qrowbase + k0];
            qf[kk][2 * v + 1] = Q[qrowbase + k0 + 1];
        }
    }

    v8f oacc[8];
    #pragma unroll
    for (int t = 0; t < 8; ++t) oacc[t] = v8f_zero();
    float mrow[8], lrow[8];
    #pragma unroll
    for (int v = 0; v < 8; ++v) { mrow[v] = -3.0e38f; lrow[v] = 0.f; }

    const int nkv = (qblk + 1) * 2;   // causal: only KV blocks up to the diagonal
    for (int kb = 0; kb < nkv; ++kb) {
        const int kv0 = kb * BKV;
        // async-stage K and V tiles (64 x 128 each), no VGPR round-trip
        #pragma unroll
        for (int it = 0; it < 4; ++it) {
            int idx = tid + it * 256;
            int r = idx >> 4;
            int c = (idx & 15) << 3;
            const __bf16* kp = KV + (size_t)(b * S + kv0 + r) * 4096 + h * 256 + c;
            async_copy_b128(kp,       &Ks[r][c]);
            async_copy_b128(kp + 128, &Vs[r][c]);
        }
        async_wait0();
        __syncthreads();

        // S = Q K^T : per-wave 16x64 (4 N-subtiles), 4 K-steps
        v8f sf[4];
        sf[0] = v8f_zero(); sf[1] = v8f_zero(); sf[2] = v8f_zero(); sf[3] = v8f_zero();
        #pragma unroll
        for (int kk = 0; kk < 4; ++kk) {
            #pragma unroll
            for (int j = 0; j < 4; ++j) {
                v16bf bfr;
                #pragma unroll
                for (int v = 0; v < 8; ++v) {
                    int k = kk * 32 + ((v < 4) ? (2 * v + 8 * lh) : (16 + 2 * (v - 4) + 8 * lh));
                    bfr[2 * v]     = Ks[j * 16 + lr][k];
                    bfr[2 * v + 1] = Ks[j * 16 + lr][k + 1];
                }
                sf[j] = __builtin_amdgcn_wmma_f32_16x16x32_bf16(
                    false, qf[kk], false, bfr, (short)0, sf[j], false, false);
            }
        }

        // causal mask + scale + online softmax (row owners: 16 lanes sharing lh)
        float alpha[8];
        #pragma unroll
        for (int v = 0; v < 8; ++v) {
            int qrow = qbase + mbase + v + 8 * lh;
            float mx = mrow[v];
            #pragma unroll
            for (int j = 0; j < 4; ++j) {
                int krow = kv0 + j * 16 + lr;
                float s = (krow <= qrow) ? sf[j][v] * scaling : -3.0e38f;
                sf[j][v] = s;
                mx = fmaxf(mx, s);
            }
            mx = fmaxf(mx, __shfl_xor(mx, 1, 32));
            mx = fmaxf(mx, __shfl_xor(mx, 2, 32));
            mx = fmaxf(mx, __shfl_xor(mx, 4, 32));
            mx = fmaxf(mx, __shfl_xor(mx, 8, 32));
            alpha[v] = __expf(mrow[v] - mx);
            float rsum = 0.f;
            #pragma unroll
            for (int j = 0; j < 4; ++j) {
                float p = __expf(sf[j][v] - mx);
                sf[j][v] = p;
                rsum += p;
            }
            rsum += __shfl_xor(rsum, 1, 32);
            rsum += __shfl_xor(rsum, 2, 32);
            rsum += __shfl_xor(rsum, 4, 32);
            rsum += __shfl_xor(rsum, 8, 32);
            lrow[v] = alpha[v] * lrow[v] + rsum;
            mrow[v] = mx;
        }

        // rescale running output
        #pragma unroll
        for (int t = 0; t < 8; ++t)
            #pragma unroll
            for (int v = 0; v < 8; ++v)
                oacc[t][v] *= alpha[v];

        // stash P (C-layout) into wave-private LDS to reload in A-layout
        #pragma unroll
        for (int j = 0; j < 4; ++j)
            #pragma unroll
            for (int v = 0; v < 8; ++v)
                Ps[wave][v + 8 * lh][j * 16 + lr] = f2bf(sf[j][v]);

        // O += P @ V : contraction over n=64 (2 K-steps), 8 d-subtiles
        #pragma unroll
        for (int kk = 0; kk < 2; ++kk) {
            v16bf af;
            #pragma unroll
            for (int v = 0; v < 8; ++v) {
                int k0 = kk * 32 + ((v < 4) ? (2 * v + 8 * lh) : (16 + 2 * (v - 4) + 8 * lh));
                af[2 * v]     = Ps[wave][lr][k0];
                af[2 * v + 1] = Ps[wave][lr][k0 + 1];
            }
            #pragma unroll
            for (int t = 0; t < 8; ++t) {
                v16bf bfr;
                #pragma unroll
                for (int v = 0; v < 8; ++v) {
                    int k = kk * 32 + ((v < 4) ? (2 * v + 8 * lh) : (16 + 2 * (v - 4) + 8 * lh));
                    bfr[2 * v]     = Vs[k][t * 16 + lr];
                    bfr[2 * v + 1] = Vs[k + 1][t * 16 + lr];
                }
                oacc[t] = __builtin_amdgcn_wmma_f32_16x16x32_bf16(
                    false, af, false, bfr, (short)0, oacc[t], false, false);
            }
        }
        __syncthreads();
    }

    // normalize and store attention output (bf16)
    #pragma unroll
    for (int v = 0; v < 8; ++v) {
        float inv = 1.0f / lrow[v];
        size_t orow = (size_t)(b * S + qbase + mbase + v + 8 * lh) * 2048 + h * 128;
        #pragma unroll
        for (int t = 0; t < 8; ++t)
            Oa[orow + t * 16 + lr] = f2bf(oacc[t][v] * inv);
    }
}

// ---------------------------------------------------------------------------
// Host-side orchestration
// ---------------------------------------------------------------------------
extern "C" void kernel_launch(void* const* d_in, const int* in_sizes, int n_in,
                              void* d_out, int out_size, void* d_ws, size_t ws_size,
                              hipStream_t stream) {
    (void)in_sizes; (void)n_in; (void)out_size; (void)ws_size;
    constexpr int B = 2, S = 2048, H = 2048, NH = 16;
    constexpr int BS = B * S;                 // 4096 rows

    const int*   positions = (const int*)d_in[0];
    const float* hs        = (const float*)d_in[1];   // (B,S,H)
    const float* wq        = (const float*)d_in[2];   // (H, 3072)
    const float* wkv_a     = (const float*)d_in[3];   // (H, 576)
    const float* ln_w      = (const float*)d_in[4];   // (512,)
    const float* wkv_b     = (const float*)d_in[5];   // (512, 4096)
    const float* wo        = (const float*)d_in[6];   // (2048, 2048)

    float* out = (float*)d_out;                       // (B,S,H)
    float* kvf = out + (size_t)BS * H;                // (B,S,576)

    char* ws = (char*)d_ws;
    auto alloc = [&](size_t bytes) {
        char* p = ws;
        ws += (bytes + 255) & ~(size_t)255;
        return p;
    };
    __bf16* hs_bf    = (__bf16*)alloc((size_t)BS * H * 2);        // 16.8 MB
    __bf16* wqn_bf   = (__bf16*)alloc((size_t)H * 2048 * 2);      //  8.4 MB
    __bf16* wkva_bf  = (__bf16*)alloc((size_t)H * 576 * 2);       //  2.4 MB
    __bf16* wkvb_bf  = (__bf16*)alloc((size_t)512 * 4096 * 2);    //  4.2 MB
    __bf16* wo_bf    = (__bf16*)alloc((size_t)2048 * 2048 * 2);   //  8.4 MB
    __bf16* q_bf     = (__bf16*)alloc((size_t)BS * 2048 * 2);     // 16.8 MB
    float*  ckv_f32  = (float*) alloc((size_t)BS * 576 * 4);      //  9.4 MB
    __bf16* comp_bf  = (__bf16*)alloc((size_t)BS * 512 * 2);      //  4.2 MB
    __bf16* kv_bf    = (__bf16*)alloc((size_t)BS * 4096 * 2);     // 33.6 MB
    __bf16* attn_bf  = (__bf16*)alloc((size_t)BS * 2048 * 2);     // 16.8 MB

    // 1) convert weights & activations to bf16
    f32_to_bf16_kernel<<<(BS * H) / 256, 256, 0, stream>>>(hs, hs_bf, BS * H);
    wq_gather_bf16_kernel<<<(H * 2048) / 256, 256, 0, stream>>>(wq, wqn_bf);
    f32_to_bf16_kernel<<<(H * 576) / 256, 256, 0, stream>>>(wkv_a, wkva_bf, H * 576);
    f32_to_bf16_kernel<<<(512 * 4096) / 256, 256, 0, stream>>>(wkv_b, wkvb_bf, 512 * 4096);
    f32_to_bf16_kernel<<<(2048 * 2048) / 256, 256, 0, stream>>>(wo, wo_bf, 2048 * 2048);

    // 2) q_nope = hs @ wq_nope  (4096x2048x2048) -> bf16
    gemm_bf16_kernel<128, true><<<dim3(2048 / 128, BS / 128), 256, 0, stream>>>(
        hs_bf, wqn_bf, q_bf, H, H, 2048, 2048);

    // 3) ckv = hs @ wkv_a       (4096x576x2048) -> f32
    gemm_bf16_kernel<64, false><<<dim3(576 / 64, BS / 128), 256, 0, stream>>>(
        hs_bf, wkva_bf, ckv_f32, H, H, 576, 576);

    // 4) rmsnorm + rope -> kv_fused (f32 out) and compressed (bf16)
    rmsnorm_rope_kernel<<<BS, 256, 0, stream>>>(ckv_f32, ln_w, positions, kvf, comp_bf);

    // 5) kv = compressed @ wkv_b  (4096x4096x512) -> bf16 (k_nope | v per head)
    gemm_bf16_kernel<128, true><<<dim3(4096 / 128, BS / 128), 256, 0, stream>>>(
        comp_bf, wkvb_bf, kv_bf, 512, 512, 4096, 4096);

    // 6) causal flash attention per (b, h, 128-row q block)
    mla_attn_kernel<<<dim3(S / 128, B * NH), 256, 0, stream>>>(q_bf, kv_bf, attn_bf);

    // 7) output = attn @ wo  (4096x2048x2048) -> f32
    gemm_bf16_kernel<128, false><<<dim3(2048 / 128, BS / 128), 256, 0, stream>>>(
        attn_bf, wo_bf, out, 2048, 2048, 2048, 2048);
}
